// TransformerBlock_81492709474980
// MI455X (gfx1250) — matrix operand order repaired and checked by reading, not compiled, not run on hardware
//
#include <hip/hip_runtime.h>
#include <hip/hip_bf16.h>
#include <stdint.h>

// ---------------------------------------------------------------------------
// Transformer block for MI455X (gfx1250, wave32, WMMA).
// Matmuls via v_wmma_f32_16x16x32_bf16 (fp32 accumulate).
// Global->LDS staging via GLOBAL_LOAD_ASYNC_TO_LDS_B128, double-buffered.
// ---------------------------------------------------------------------------

#define DEV __device__ __forceinline__

typedef __attribute__((ext_vector_type(16))) __bf16 v16bf;
typedef __attribute__((ext_vector_type(8)))  float  v8f;
typedef __attribute__((ext_vector_type(4)))  int    v4i_t;

static constexpr int BATCH = 2;
static constexpr int SEQ   = 2048;
static constexpr int EMB   = 1024;
static constexpr int FFD   = 4096;
static constexpr int NHEAD = 16;
static constexpr int HDIM  = 64;
static constexpr int NTOK  = BATCH * SEQ;   // 4096

// ---------------- async global -> LDS copy --------------------------------

#if __has_builtin(__builtin_amdgcn_global_load_async_to_lds_b128)
#define HAS_ASYNC_LDS 1
#endif

DEV void cp_async_b128(__bf16* lds, const __bf16* g) {
#ifdef HAS_ASYNC_LDS
    __builtin_amdgcn_global_load_async_to_lds_b128(
        (__attribute__((address_space(1))) v4i_t*)(__bf16*)g,
        (__attribute__((address_space(3))) v4i_t*)lds,
        0, 0);
#else
    *reinterpret_cast<uint4*>(lds) = *reinterpret_cast<const uint4*>(g);
#endif
}

DEV void wait_async() {
#ifdef HAS_ASYNC_LDS
#if __has_builtin(__builtin_amdgcn_s_wait_asynccnt)
    __builtin_amdgcn_s_wait_asynccnt(0);
#else
    asm volatile("s_wait_asynccnt 0" ::: "memory");
#endif
#endif
}

// ---------------- WMMA helpers -------------------------------------------

DEV v8f wmma_bf16(v16bf a, v16bf b, v8f c) {
    return __builtin_amdgcn_wmma_f32_16x16x32_bf16(
        /*neg_a=*/false, a, /*neg_b=*/false, b,
        /*c_mod=*/(short)0, c, /*reuse_a=*/false, /*reuse_b=*/false);
}

// A-fragment (16x32, MxK): lanes 0-15 row M=l16 hold K = [0..7],[16..23];
// lanes 16-31 hold K = [8..15],[24..31]. Two 16B chunks.
DEV v16bf load_frag_a(const __bf16* base, int row, int stride, int k0, int hf) {
    const __bf16* p = base + row * stride + k0 + hf * 8;
    union { uint4 u[2]; v16bf v; } f;
    f.u[0] = *reinterpret_cast<const uint4*>(p);
    f.u[1] = *reinterpret_cast<const uint4*>(p + 16);
    return f.v;
}

// B-fragment (32x16, KxN) with B stored K-contiguous per column (Bt[N][K]):
// lanes 0-15 col N=l16 hold K = [0..7],[16..23]; lanes 16-31 hold K = [8..15],[24..31].
DEV v16bf load_frag_b(const __bf16* base, int row, int stride, int k0, int hf) {
    const __bf16* p = base + row * stride + k0 + hf * 8;
    union { uint4 u[2]; v16bf v; } f;
    f.u[0] = *reinterpret_cast<const uint4*>(p);
    f.u[1] = *reinterpret_cast<const uint4*>(p + 16);
    return f.v;
}

DEV float gelu_f(float x) {
    const float c = 0.7978845608028654f;
    return 0.5f * x * (1.0f + tanhf(c * (x + 0.044715f * x * x * x)));
}

// ---------------- weight transpose + fp32->bf16 convert -------------------
// W: [K][N] fp32 row-major  ->  Wt: [N][K] bf16 row-major
__global__ void transpose_convert_kernel(const float* __restrict__ W,
                                         __bf16* __restrict__ Wt,
                                         int K, int N) {
    __shared__ float tile[32][33];
    int bx = blockIdx.x, by = blockIdx.y;
    int tx = threadIdx.x, ty = threadIdx.y;
    #pragma unroll
    for (int i = 0; i < 4; ++i) {
        int r = by * 32 + ty + i * 8;
        int c = bx * 32 + tx;
        tile[ty + i * 8][tx] = W[(size_t)r * N + c];
    }
    __syncthreads();
    #pragma unroll
    for (int i = 0; i < 4; ++i) {
        int n = bx * 32 + ty + i * 8;
        int k = by * 32 + tx;
        Wt[(size_t)n * K + k] = (__bf16)tile[tx][ty + i * 8];
    }
}

// ---------------- LayerNorm (fp32 in -> bf16 out) -------------------------
__global__ void layernorm_kernel(const float* __restrict__ x,
                                 const float* __restrict__ g,
                                 const float* __restrict__ be,
                                 __bf16* __restrict__ out) {
    __shared__ float s1[256];
    __shared__ float s2[256];
    int tok = blockIdx.x;
    int tid = threadIdx.x;
    const float4 v = reinterpret_cast<const float4*>(x + (size_t)tok * EMB)[tid];
    float sum = v.x + v.y + v.z + v.w;
    float sq  = v.x * v.x + v.y * v.y + v.z * v.z + v.w * v.w;
    s1[tid] = sum; s2[tid] = sq;
    __syncthreads();
    for (int s = 128; s > 0; s >>= 1) {
        if (tid < s) { s1[tid] += s1[tid + s]; s2[tid] += s2[tid + s]; }
        __syncthreads();
    }
    float mu  = s1[0] * (1.0f / EMB);
    float var = s2[0] * (1.0f / EMB) - mu * mu;
    float rs  = rsqrtf(var + 1e-5f);
    float vv[4] = {v.x, v.y, v.z, v.w};
    #pragma unroll
    for (int j = 0; j < 4; ++j) {
        int c = tid * 4 + j;
        out[(size_t)tok * EMB + c] = (__bf16)((vv[j] - mu) * rs * g[c] + be[c]);
    }
}

// ---------------- bf16 WMMA GEMM (double-buffered LDS) --------------------
// C[M,N] = A[M,K] @ Wt[N,K]^T, 128x128 block, 8 waves, wave tile 64x32.
// MODE 0: store bf16 into [B,H,S,Hd]  (Q, K projections; oscale folds 1/sqrt(Hd))
// MODE 1: store bf16 into [B,H,Hd,S]  (V projection, transposed)
// MODE 2: store fp32 = acc + bias + residual
// MODE 3: store bf16 = gelu(acc + bias)
template <int MODE>
__global__ void gemm_bf16_kernel(const __bf16* __restrict__ A,
                                 const __bf16* __restrict__ Bt,
                                 const float* __restrict__ bias,
                                 const float* __restrict__ res,
                                 void* __restrict__ outp,
                                 int M, int N, int K, float oscale) {
    constexpr int LDT = 40;                 // 32 + 8 pad (bf16)
    __shared__ __bf16 sa[2][128 * LDT];
    __shared__ __bf16 sb[2][128 * LDT];

    const int tid  = threadIdx.x;
    const int lane = tid & 31;
    const int wave = tid >> 5;
    const int hf   = lane >> 4;
    const int l16  = lane & 15;
    const int wm   = (wave & 1) * 64;
    const int wn   = (wave >> 1) * 32;
    const int bM   = blockIdx.y * 128;
    const int bN   = blockIdx.x * 128;

    // staging geometry: 512 16B chunks per tile, 2 per thread
    const int srow0 = tid >> 2;             // rows 0..63
    const int sc8   = (tid & 3) * 8;
    const __bf16* Ag = A  + (size_t)(bM + srow0) * K + sc8;
    const __bf16* Bg = Bt + (size_t)(bN + srow0) * K + sc8;

    auto stage = [&](int buf, int k0) {
        #pragma unroll
        for (int i = 0; i < 2; ++i) {
            int row = srow0 + i * 64;
            cp_async_b128(&sa[buf][row * LDT + sc8], Ag + (size_t)i * 64 * K + k0);
            cp_async_b128(&sb[buf][row * LDT + sc8], Bg + (size_t)i * 64 * K + k0);
        }
    };

    v8f acc[4][2];
    #pragma unroll
    for (int m = 0; m < 4; ++m)
        #pragma unroll
        for (int n = 0; n < 2; ++n)
            acc[m][n] = (v8f){0.f,0.f,0.f,0.f,0.f,0.f,0.f,0.f};

    const int KT = K >> 5;
    stage(0, 0);
    for (int kt = 0; kt < KT; ++kt) {
        const int cur = kt & 1;
        wait_async();
        __syncthreads();                    // tile[cur] resident; tile[cur^1] free
        if (kt + 1 < KT) stage(cur ^ 1, (kt + 1) << 5);

        v16bf bfr[2];
        #pragma unroll
        for (int n = 0; n < 2; ++n)
            bfr[n] = load_frag_b(sb[cur], wn + n * 16 + l16, LDT, 0, hf);
        #pragma unroll
        for (int m = 0; m < 4; ++m) {
            v16bf afr = load_frag_a(sa[cur], wm + m * 16 + l16, LDT, 0, hf);
            #pragma unroll
            for (int n = 0; n < 2; ++n)
                acc[m][n] = wmma_bf16(afr, bfr[n], acc[m][n]);
        }
    }

    // epilogue: C layout row = e + 8*hf, col = l16
    #pragma unroll
    for (int m = 0; m < 4; ++m) {
        #pragma unroll
        for (int n = 0; n < 2; ++n) {
            #pragma unroll
            for (int e = 0; e < 8; ++e) {
                int grow = bM + wm + m * 16 + e + 8 * hf;
                int gcol = bN + wn + n * 16 + l16;
                float vres = (acc[m][n][e] + bias[gcol]) * oscale;
                if constexpr (MODE == 0) {          // [B,H,S,Hd] bf16
                    int b = grow >> 11, s = grow & 2047;
                    int h = gcol >> 6,  d = gcol & 63;
                    ((__bf16*)outp)[(((size_t)(b * NHEAD + h)) * SEQ + s) * HDIM + d] =
                        (__bf16)vres;
                } else if constexpr (MODE == 1) {   // [B,H,Hd,S] bf16
                    int b = grow >> 11, s = grow & 2047;
                    int h = gcol >> 6,  d = gcol & 63;
                    ((__bf16*)outp)[(((size_t)(b * NHEAD + h)) * HDIM + d) * SEQ + s] =
                        (__bf16)vres;
                } else if constexpr (MODE == 2) {   // fp32 + residual
                    size_t idx = (size_t)grow * N + gcol;
                    ((float*)outp)[idx] = vres + res[idx];
                } else {                            // gelu -> bf16
                    ((__bf16*)outp)[(size_t)grow * N + gcol] = (__bf16)gelu_f(vres);
                }
            }
        }
    }
}

// ---------------- Flash attention (double-buffered K/V) -------------------
// grid: (SEQ/64, BATCH*NHEAD), block: 128 (4 waves, 16 queries each)
// Q,K: [B,H,S,Hd] bf16 (Q pre-scaled by 1/sqrt(Hd)) ; Vt: [B,H,Hd,S] bf16
// out: [NTOK, EMB] bf16
__global__ void attention_kernel(const __bf16* __restrict__ Qg,
                                 const __bf16* __restrict__ Kg,
                                 const __bf16* __restrict__ Vt,
                                 __bf16* __restrict__ attO) {
    constexpr int LDT = 72;                 // 64 + 8 pad
    __shared__ __bf16 qs[64 * LDT];
    __shared__ __bf16 ks[2][64 * LDT];
    __shared__ __bf16 vs[2][64 * LDT];
    __shared__ __bf16 ps[4 * 16 * LDT];

    const int tid  = threadIdx.x;
    const int lane = tid & 31;
    const int wave = tid >> 5;
    const int hf   = lane >> 4;
    const int l16  = lane & 15;
    const int qbase = blockIdx.x * 64;
    const int bh    = blockIdx.y;           // b*16 + h
    const int b     = bh >> 4;
    const int h     = bh & 15;

    // staging geometry: 512 chunks per 64x64 tile, 4 per thread
    const int srow0 = tid >> 3;             // rows 0..15
    const int sc8   = (tid & 7) * 8;
    const __bf16* Qp = Qg + ((size_t)bh * SEQ + qbase) * HDIM + (size_t)srow0 * HDIM + sc8;
    const __bf16* Kp = Kg + ((size_t)bh * SEQ) * HDIM + (size_t)srow0 * HDIM + sc8;
    const __bf16* Vp = Vt + ((size_t)bh * HDIM + srow0) * SEQ + sc8;

    auto stageKV = [&](int buf, int kt) {
        #pragma unroll
        for (int i = 0; i < 4; ++i) {
            cp_async_b128(&ks[buf][(srow0 + i * 16) * LDT + sc8],
                          Kp + ((size_t)kt * 64 + i * 16) * HDIM);
            cp_async_b128(&vs[buf][(srow0 + i * 16) * LDT + sc8],
                          Vp + (size_t)i * 16 * SEQ + kt * 64);
        }
    };

    // stage Q tile (64x64) and first K/V tile together
    #pragma unroll
    for (int i = 0; i < 4; ++i)
        cp_async_b128(&qs[(srow0 + i * 16) * LDT + sc8], Qp + (size_t)i * 16 * HDIM);
    stageKV(0, 0);
    wait_async();
    __syncthreads();

    v8f   o[4];
    float mrow[8], lrow[8];
    #pragma unroll
    for (int n = 0; n < 4; ++n) o[n] = (v8f){0.f,0.f,0.f,0.f,0.f,0.f,0.f,0.f};
    #pragma unroll
    for (int e = 0; e < 8; ++e) { mrow[e] = -3.0e38f; lrow[e] = 0.f; }

    constexpr int KT = SEQ / 64;
    for (int kt = 0; kt < KT; ++kt) {
        const int cur = kt & 1;
        if (kt + 1 < KT) stageKV(cur ^ 1, kt + 1);   // overlap DMA with compute

        // Q fragments re-read from resident LDS tile each iteration (cheaper
        // than letting the allocator spill a register-resident copy to scratch)
        v16bf qf0 = load_frag_a(qs, wave * 16 + l16, LDT, 0,  hf);
        v16bf qf1 = load_frag_a(qs, wave * 16 + l16, LDT, 32, hf);

        // scores: S = Qs @ K^T  (4 key sub-tiles of 16); Q carries 1/sqrt(Hd)
        v8f sc[4];
        #pragma unroll
        for (int n = 0; n < 4; ++n) {
            sc[n] = (v8f){0.f,0.f,0.f,0.f,0.f,0.f,0.f,0.f};
            v16bf kb0 = load_frag_b(ks[cur], n * 16 + l16, LDT, 0,  hf);
            v16bf kb1 = load_frag_b(ks[cur], n * 16 + l16, LDT, 32, hf);
            sc[n] = wmma_bf16(qf0, kb0, sc[n]);
            sc[n] = wmma_bf16(qf1, kb1, sc[n]);
        }

        // online softmax per row (row = e + 8*hf; cols span lanes in 16-group)
        #pragma unroll
        for (int e = 0; e < 8; ++e) {
            float rm = -3.0e38f;
            #pragma unroll
            for (int n = 0; n < 4; ++n) rm = fmaxf(rm, sc[n][e]);
            #pragma unroll
            for (int m = 1; m < 16; m <<= 1)
                rm = fmaxf(rm, __shfl_xor(rm, m, 32));
            float mn  = fmaxf(mrow[e], rm);
            float scl = __expf(mrow[e] - mn);
            mrow[e]   = mn;
            float rs  = 0.f;
            int rloc  = e + 8 * hf;
            #pragma unroll
            for (int n = 0; n < 4; ++n) {
                float p = __expf(sc[n][e] - mn);
                ps[(wave * 16 + rloc) * LDT + n * 16 + l16] = (__bf16)p;
                rs += p;
            }
            #pragma unroll
            for (int m = 1; m < 16; m <<= 1)
                rs += __shfl_xor(rs, m, 32);
            lrow[e] = lrow[e] * scl + rs;
            #pragma unroll
            for (int n = 0; n < 4; ++n) o[n][e] *= scl;
        }
        // no barrier: each wave reads back only its own ps rows; LDS ops from
        // the same wave complete in order (DScnt), compiler inserts the wait.

        // O += P @ V  (4 d sub-tiles of 16)
        const __bf16* pw = ps + wave * 16 * LDT;
        v16bf pf0 = load_frag_a(pw, l16, LDT, 0,  hf);
        v16bf pf1 = load_frag_a(pw, l16, LDT, 32, hf);
        #pragma unroll
        for (int n = 0; n < 4; ++n) {
            v16bf vb0 = load_frag_b(vs[cur], n * 16 + l16, LDT, 0,  hf);
            v16bf vb1 = load_frag_b(vs[cur], n * 16 + l16, LDT, 32, hf);
            o[n] = wmma_bf16(pf0, vb0, o[n]);
            o[n] = wmma_bf16(pf1, vb1, o[n]);
        }

        if (kt + 1 < KT) {
            wait_async();                   // next tile resident (own loads) ...
            __syncthreads();                // ... and across all waves
        }
    }

    // write out: token-major bf16 [NTOK, EMB]
    #pragma unroll
    for (int e = 0; e < 8; ++e) {
        int rloc  = e + 8 * hf;
        int tok   = b * SEQ + qbase + wave * 16 + rloc;
        float inv = 1.0f / lrow[e];
        #pragma unroll
        for (int n = 0; n < 4; ++n) {
            int col = h * HDIM + n * 16 + l16;
            attO[(size_t)tok * EMB + col] = (__bf16)(o[n][e] * inv);
        }
    }
}

// ---------------------------------------------------------------------------

extern "C" void kernel_launch(void* const* d_in, const int* in_sizes, int n_in,
                              void* d_out, int out_size, void* d_ws, size_t ws_size,
                              hipStream_t stream) {
    const float* x   = (const float*)d_in[0];
    const float* Wq  = (const float*)d_in[1];
    const float* bq  = (const float*)d_in[2];
    const float* Wk  = (const float*)d_in[3];
    const float* bk  = (const float*)d_in[4];
    const float* Wv  = (const float*)d_in[5];
    const float* bv  = (const float*)d_in[6];
    const float* Wp  = (const float*)d_in[7];
    const float* bp  = (const float*)d_in[8];
    const float* W1  = (const float*)d_in[9];
    const float* b1  = (const float*)d_in[10];
    const float* W2  = (const float*)d_in[11];
    const float* b2  = (const float*)d_in[12];
    const float* g1  = (const float*)d_in[13];
    const float* be1 = (const float*)d_in[14];
    const float* g2  = (const float*)d_in[15];
    const float* be2 = (const float*)d_in[16];

    uint8_t* ws = (uint8_t*)d_ws;
    size_t off = 0;
    auto carve = [&](size_t bytes) -> void* {
        void* p = ws + off;
        off += (bytes + 255) & ~(size_t)255;
        return p;
    };
    __bf16* Wqt  = (__bf16*)carve((size_t)EMB * EMB * 2);
    __bf16* Wkt  = (__bf16*)carve((size_t)EMB * EMB * 2);
    __bf16* Wvt  = (__bf16*)carve((size_t)EMB * EMB * 2);
    __bf16* Wpt  = (__bf16*)carve((size_t)EMB * EMB * 2);
    __bf16* W1t  = (__bf16*)carve((size_t)FFD * EMB * 2);
    __bf16* W2t  = (__bf16*)carve((size_t)EMB * FFD * 2);
    __bf16* hbf  = (__bf16*)carve((size_t)NTOK * EMB * 2);
    __bf16* Qb   = (__bf16*)carve((size_t)NTOK * EMB * 2);
    __bf16* Kb   = (__bf16*)carve((size_t)NTOK * EMB * 2);
    __bf16* Vtb  = (__bf16*)carve((size_t)NTOK * EMB * 2);
    __bf16* attO = (__bf16*)carve((size_t)NTOK * EMB * 2);
    float*  x1f  = (float*)carve((size_t)NTOK * EMB * 4);
    __bf16* h2b  = (__bf16*)carve((size_t)NTOK * EMB * 2);
    __bf16* act  = (__bf16*)carve((size_t)NTOK * FFD * 2);
    (void)ws_size; (void)in_sizes; (void)n_in; (void)out_size;

    dim3 tb(32, 8);
    transpose_convert_kernel<<<dim3(EMB / 32, EMB / 32), tb, 0, stream>>>(Wq, Wqt, EMB, EMB);
    transpose_convert_kernel<<<dim3(EMB / 32, EMB / 32), tb, 0, stream>>>(Wk, Wkt, EMB, EMB);
    transpose_convert_kernel<<<dim3(EMB / 32, EMB / 32), tb, 0, stream>>>(Wv, Wvt, EMB, EMB);
    transpose_convert_kernel<<<dim3(EMB / 32, EMB / 32), tb, 0, stream>>>(Wp, Wpt, EMB, EMB);
    transpose_convert_kernel<<<dim3(FFD / 32, EMB / 32), tb, 0, stream>>>(W1, W1t, EMB, FFD);
    transpose_convert_kernel<<<dim3(EMB / 32, FFD / 32), tb, 0, stream>>>(W2, W2t, FFD, EMB);

    layernorm_kernel<<<NTOK, 256, 0, stream>>>(x, g1, be1, hbf);

    const float qscale = 0.125f;            // 1/sqrt(HDIM)
    dim3 gProj(EMB / 128, NTOK / 128);      // (8, 32)
    gemm_bf16_kernel<0><<<gProj, 256, 0, stream>>>(hbf, Wqt, bq, nullptr, Qb, NTOK, EMB, EMB, qscale);
    gemm_bf16_kernel<0><<<gProj, 256, 0, stream>>>(hbf, Wkt, bk, nullptr, Kb, NTOK, EMB, EMB, 1.0f);
    gemm_bf16_kernel<1><<<gProj, 256, 0, stream>>>(hbf, Wvt, bv, nullptr, Vtb, NTOK, EMB, EMB, 1.0f);

    attention_kernel<<<dim3(SEQ / 64, BATCH * NHEAD), 128, 0, stream>>>(Qb, Kb, Vtb, attO);

    gemm_bf16_kernel<2><<<gProj, 256, 0, stream>>>(attO, Wpt, bp, x, x1f, NTOK, EMB, EMB, 1.0f);

    layernorm_kernel<<<NTOK, 256, 0, stream>>>(x1f, g2, be2, h2b);

    dim3 gFF1(FFD / 128, NTOK / 128);       // (32, 32)
    gemm_bf16_kernel<3><<<gFF1, 256, 0, stream>>>(h2b, W1t, b1, nullptr, act, NTOK, FFD, EMB, 1.0f);

    gemm_bf16_kernel<2><<<gProj, 256, 0, stream>>>(act, W2t, b2, x1f, d_out, NTOK, EMB, FFD, 1.0f);
}
